// VectorQuantizerCosine_11166914969844
// MI455X (gfx1250) — matrix operand order repaired and checked by reading, not compiled, not run on hardware
//
#include <hip/hip_runtime.h>
#include <math.h>

// ---- CDNA5 WMMA vector types ----
typedef __attribute__((ext_vector_type(16))) __bf16 v16bf;
typedef __attribute__((ext_vector_type(8)))  float  v8f;

#define N_CODES 8192
#define C_DIM   256
#define N_ROWS  8192          // 8*32*32
#define KB      8             // 256 / 32
#define OUT_ZQ   2097152
#define OUT_LOSS (OUT_ZQ)
#define OUT_PERP (OUT_ZQ + 1)
#define OUT_IDX  (OUT_ZQ + 2)

__device__ __forceinline__ unsigned short f2bf(float f) {
  unsigned u = __float_as_uint(f);
  unsigned r = u + 0x7FFFu + ((u >> 16) & 1u);   // round-to-nearest-even
  return (unsigned short)(r >> 16);
}
__device__ __forceinline__ float bf2f(unsigned short h) {
  return __uint_as_float(((unsigned)h) << 16);
}

// ---- zero the accumulators (hist + loss) each call: deterministic ----
__global__ void vq_init(float* __restrict__ hist, float* __restrict__ loss) {
  int i = blockIdx.x * blockDim.x + threadIdx.x;
  if (i < N_CODES) hist[i] = 0.0f;
  if (i == 0) loss[0] = 0.0f;
}

// ---- pack z (NCHW f32) into WMMA A-operand layout, split hi/lo bf16 ----
// A 16x32 bf16 k order chosen here (contiguous): lane<16: M=lane, slots 0..15 -> K=0..15 ;
// lane>=16: M=lane-16, K +16 (matches B packing).
__global__ void vq_pack_a(const float* __restrict__ z,
                          unsigned short* __restrict__ a_hi,
                          unsigned short* __restrict__ a_lo) {
  unsigned gid  = blockIdx.x * blockDim.x + threadIdx.x;  // 2^21 threads
  unsigned slot = gid & 15u;
  unsigned lane = (gid >> 4) & 31u;
  unsigned kb   = (gid >> 9) & 7u;
  unsigned nb   = gid >> 12;
  unsigned K = slot + (lane < 16u ? 0u : 16u);
  unsigned M = lane & 15u;
  unsigned n = nb * 16u + M;                 // flattened (b,h,w)
  unsigned c = kb * 32u + K;
  float f = z[(size_t)(n >> 10) * 262144u + (size_t)c * 1024u + (n & 1023u)];
  unsigned short hi = f2bf(f);
  float lo = f - bf2f(hi);
  a_hi[gid] = hi;
  a_lo[gid] = f2bf(lo);
}

// ---- pack embed into WMMA B-operand layout (B = embed^T, 32x16), hi/lo ----
// B 32x16 bf16 k order chosen here (contiguous): lane<16: N=lane, slots 0..15 -> K=0..15 ; lane>=16: K +16.
__global__ void vq_pack_b(const float* __restrict__ embed,
                          unsigned short* __restrict__ b_hi,
                          unsigned short* __restrict__ b_lo) {
  unsigned gid  = blockIdx.x * blockDim.x + threadIdx.x;  // 2^21 threads
  unsigned slot = gid & 15u;
  unsigned lane = (gid >> 4) & 31u;
  unsigned kb   = (gid >> 9) & 7u;
  unsigned eb   = gid >> 12;
  unsigned K = slot + (lane < 16u ? 0u : 16u);
  unsigned N = lane & 15u;
  unsigned j = eb * 16u + N;
  float f = embed[(size_t)j * C_DIM + kb * 32u + K];
  unsigned short hi = f2bf(f);
  float lo = f - bf2f(hi);
  b_hi[gid] = hi;
  b_lo[gid] = f2bf(lo);
}

// ---- 0.5 * ||e_j||^2 per code ----
__global__ void vq_ehalf(const float* __restrict__ embed, float* __restrict__ ehalf) {
  int j = blockIdx.x * blockDim.x + threadIdx.x;
  if (j >= N_CODES) return;
  const float4* e4 = (const float4*)(embed + (size_t)j * C_DIM);
  float s = 0.0f;
#pragma unroll 8
  for (int i = 0; i < C_DIM / 4; ++i) {
    float4 v = e4[i];
    s += v.x * v.x + v.y * v.y + v.z * v.z + v.w * v.w;
  }
  ehalf[j] = 0.5f * s;
}

// ---- fused GEMM (bf16-split WMMA, fp32-accurate) + argmax over codes ----
// argmin_j ||z-e_j||^2  ==  argmax_j ( z.e_j - 0.5*||e_j||^2 )
// Blocking for L2/WGP$ efficiency: the 8 waves of a block each own a
// DIFFERENT 16-row A-tile (kept in registers) and stream the SAME B range,
// so WGP$ broadcasts each B cacheline to all 8 waves (one L2 fill per WGP).
// grid = 64 M-groups x 4 code-quarters; per-row candidates reduced later.
__launch_bounds__(256, 1)
__global__ void vq_argmin_wmma(const unsigned short* __restrict__ a_hi,
                               const unsigned short* __restrict__ a_lo,
                               const unsigned short* __restrict__ b_hi,
                               const unsigned short* __restrict__ b_lo,
                               const float* __restrict__ ehalf,
                               float* __restrict__ cand_val,
                               int* __restrict__ cand_idx) {
  const int mg   = blockIdx.x >> 2;        // M-group: 8 tiles of 16 rows
  const int cq   = blockIdx.x & 3;         // code quarter: 2048 codes
  const int lane = threadIdx.x & 31;
  const int wv   = threadIdx.x >> 5;
  const int mtile = mg * 8 + wv;           // this wave's 16-row tile

  // A tile (hi+lo) stays in registers for the whole scan (16 x 256, 128 VGPRs)
  v16bf ah[KB], al[KB];
#pragma unroll
  for (int kb = 0; kb < KB; ++kb) {
    size_t base = (((size_t)mtile * KB + kb) * 32 + lane) * 16;
    ah[kb] = *(const v16bf*)(a_hi + base);
    al[kb] = *(const v16bf*)(a_lo + base);
  }

  float bestv[8];
  int   besti[8];
#pragma unroll
  for (int r = 0; r < 8; ++r) { bestv[r] = -3.4e38f; besti[r] = 0; }

  const int eb0 = cq * 128;
  for (int t = 0; t < 128; ++t) {
    const int eb = eb0 + t;
    size_t tb = ((size_t)eb * (KB * 32) + lane) * 16;
    if (t + 1 < 128) {  // prefetch next B tile into near caches
      __builtin_prefetch(b_hi + tb + (size_t)KB * 32 * 16, 0, 3);
      __builtin_prefetch(b_lo + tb + (size_t)KB * 32 * 16, 0, 3);
    }
    v8f c = {};
#pragma unroll
    for (int kb = 0; kb < KB; ++kb) {
      size_t bbase = tb + (size_t)kb * 32 * 16;
      v16bf bh = *(const v16bf*)(b_hi + bbase);
      v16bf bl = *(const v16bf*)(b_lo + bbase);
      // fp32-accurate dot via bf16 split: hi*hi + hi*lo + lo*hi
      c = __builtin_amdgcn_wmma_f32_16x16x32_bf16(false, ah[kb], false, bh, (short)0, c, false, false);
      c = __builtin_amdgcn_wmma_f32_16x16x32_bf16(false, ah[kb], false, bl, (short)0, c, false, false);
      c = __builtin_amdgcn_wmma_f32_16x16x32_bf16(false, al[kb], false, bh, (short)0, c, false, false);
    }
    const int   nidx = eb * 16 + (lane & 15);
    const float eh   = ehalf[nidx];
#pragma unroll
    for (int r = 0; r < 8; ++r) {       // C layout: VGPR r -> row M=r (lanes<16) / r+8
      float s = c[r] - eh;
      if (s > bestv[r]) { bestv[r] = s; besti[r] = nidx; }
    }
    // keep the 8 waves converged so their shared B stream stays in WGP$
    if ((t & 7) == 7) __syncthreads();
  }

  // reduce across the 16 lanes (columns N) within each half-wave
#pragma unroll
  for (int r = 0; r < 8; ++r) {
    float v = bestv[r];
    int   i = besti[r];
#pragma unroll
    for (int m = 8; m >= 1; m >>= 1) {
      float ov = __shfl_xor(v, m, 16);
      int   oi = __shfl_xor(i, m, 16);
      if (ov > v || (ov == v && oi < i)) { v = ov; i = oi; }
    }
    if ((lane & 15) == 0) {
      int M   = (lane >> 4) * 8 + r;
      int row = mtile * 16 + M;
      cand_val[row * 4 + cq] = v;
      cand_idx[row * 4 + cq] = i;
    }
  }
}

// ---- pick best of the 4 per-row candidates, emit index + histogram ----
__global__ void vq_reduce_cand(const float* __restrict__ cand_val,
                               const int* __restrict__ cand_idx,
                               int* __restrict__ idx_buf,
                               float* __restrict__ out_idx,
                               float* __restrict__ hist) {
  int n = blockIdx.x * blockDim.x + threadIdx.x;
  if (n >= N_ROWS) return;
  float v = cand_val[n * 4];
  int   i = cand_idx[n * 4];
#pragma unroll
  for (int q = 1; q < 4; ++q) {
    float ov = cand_val[n * 4 + q];
    int   oi = cand_idx[n * 4 + q];
    if (ov > v || (ov == v && oi < i)) { v = ov; i = oi; }
  }
  idx_buf[n] = i;
  out_idx[n] = (float)i;
  atomicAdd(&hist[i], 1.0f);
}

// ---- gather z_q = embed[idx] back to NCHW + accumulate MSE ----
__global__ void vq_gather_loss(const float* __restrict__ z,
                               const float* __restrict__ embed,
                               const int* __restrict__ idx_buf,
                               float* __restrict__ zq,
                               float* __restrict__ loss_accum) {
  __shared__ float red[256];
  unsigned tid = blockIdx.x * blockDim.x + threadIdx.x;  // 2^21, [b][c][hw]
  unsigned b  = tid >> 18;
  unsigned c  = (tid >> 10) & 255u;
  unsigned hw = tid & 1023u;
  unsigned n  = b * 1024u + hw;
  int   j  = idx_buf[n];
  float q  = embed[(size_t)j * C_DIM + c];
  float zv = z[tid];
  zq[tid] = q;                       // z_q_st forward value == z_q
  float d = q - zv;
  red[threadIdx.x] = d * d;
  __syncthreads();
  for (int s = 128; s > 0; s >>= 1) {
    if (threadIdx.x < (unsigned)s) red[threadIdx.x] += red[threadIdx.x + s];
    __syncthreads();
  }
  if (threadIdx.x == 0) atomicAdd(loss_accum, red[0]);
}

// ---- perplexity + final loss scaling ----
__global__ void vq_finalize(const float* __restrict__ hist,
                            const float* __restrict__ loss_accum,
                            float* __restrict__ out) {
  __shared__ float red[256];
  float h = 0.0f;
  for (int i = threadIdx.x; i < N_CODES; i += 256) {
    float em = hist[i] * (1.0f / 8192.0f);
    h += em * logf(em + 1e-10f);
  }
  red[threadIdx.x] = h;
  __syncthreads();
  for (int s = 128; s > 0; s >>= 1) {
    if (threadIdx.x < (unsigned)s) red[threadIdx.x] += red[threadIdx.x + s];
    __syncthreads();
  }
  if (threadIdx.x == 0) {
    out[OUT_PERP] = expf(-red[0]);
    // loss = mean((zq-z)^2) + 0.25*mean((zq-z)^2) = 1.25 * MSE
    out[OUT_LOSS] = loss_accum[0] * 1.25f / 2097152.0f;
  }
}

extern "C" void kernel_launch(void* const* d_in, const int* in_sizes, int n_in,
                              void* d_out, int out_size, void* d_ws, size_t ws_size,
                              hipStream_t stream) {
  const float* z     = (const float*)d_in[0];   // [8,256,32,32]
  const float* embed = (const float*)d_in[1];   // [8192,256]
  float* out = (float*)d_out;
  char*  ws  = (char*)d_ws;

  unsigned short* a_hi = (unsigned short*)(ws);
  unsigned short* a_lo = (unsigned short*)(ws + (size_t)(4u << 20));
  unsigned short* b_hi = (unsigned short*)(ws + (size_t)(8u << 20));
  unsigned short* b_lo = (unsigned short*)(ws + (size_t)(12u << 20));
  float* ehalf = (float*)(ws + (size_t)(16u << 20));
  float* hist  = (float*)(ws + (size_t)(16u << 20) + 32768);
  int*   idxb  = (int*)  (ws + (size_t)(16u << 20) + 65536);
  float* lossa = (float*)(ws + (size_t)(16u << 20) + 98304);
  float* cval  = (float*)(ws + (size_t)(16u << 20) + 131072);   // 8192*4 f32
  int*   cidx  = (int*)  (ws + (size_t)(16u << 20) + 262144);   // 8192*4 i32

  vq_init       <<<32,   256, 0, stream>>>(hist, lossa);
  vq_pack_a     <<<8192, 256, 0, stream>>>(z, a_hi, a_lo);
  vq_pack_b     <<<8192, 256, 0, stream>>>(embed, b_hi, b_lo);
  vq_ehalf      <<<32,   256, 0, stream>>>(embed, ehalf);
  vq_argmin_wmma<<<256,  256, 0, stream>>>(a_hi, a_lo, b_hi, b_lo, ehalf,
                                           cval, cidx);
  vq_reduce_cand<<<32,   256, 0, stream>>>(cval, cidx, idxb, out + OUT_IDX, hist);
  vq_gather_loss<<<8192, 256, 0, stream>>>(z, embed, idxb, out, lossa);
  vq_finalize   <<<1,    256, 0, stream>>>(hist, lossa, out);
}
